// HyperbolicMuRP_69260642615717
// MI455X (gfx1250) — matrix operand order repaired and checked by reading, not compiled, not run on hardware
//
#include <hip/hip_runtime.h>
#include <hip/hip_bf16.h>
#include <stdint.h>

typedef __attribute__((ext_vector_type(16))) _Float16 v16h;
typedef __attribute__((ext_vector_type(8)))  _Float16 v8h;
typedef __attribute__((ext_vector_type(8)))  float    v8f;

#define C_CURV   0.01f
#define C_CURV2  0.0001f
#define EPS_F    1e-6f
#define NSLICES  8

// ---------------------------------------------------------------------------
// Prep: candidates f32 -> f16, plus packed per-candidate metadata:
//   cmeta[n] = { |y|^2, 1 + c|y|^2, scale*margin + bias[n], c|y|^2 }
// ---------------------------------------------------------------------------
__global__ void murp_prep_cand(const float* __restrict__ cand,
                               const float* __restrict__ bias,
                               const float* __restrict__ scaleP,
                               const float* __restrict__ marginP,
                               _Float16* __restrict__ candh,
                               float4* __restrict__ cmeta, int N) {
  int n = blockIdx.x * blockDim.x + threadIdx.x;
  if (n >= N) return;
  const float* row = cand + (size_t)n * 32;
  _Float16* out = candh + (size_t)n * 32;
  float s = 0.f;
#pragma unroll
  for (int k = 0; k < 32; ++k) {
    float v = row[k];
    s += v * v;
    out[k] = (_Float16)v;
  }
  float cys  = C_CURV * s;
  float base = scaleP[0] * marginP[0] + bias[n];
  cmeta[n] = make_float4(s, 1.f + cys, base, cys);
}

// ---------------------------------------------------------------------------
// Prep: query f32 -> f16, |q|^2, and exact f32 target logit per row
// ---------------------------------------------------------------------------
__global__ void murp_prep_query(const float* __restrict__ q,
                                const float* __restrict__ cand,
                                const int* __restrict__ target,
                                const float* __restrict__ bias,
                                const float* __restrict__ scaleP,
                                const float* __restrict__ marginP,
                                _Float16* __restrict__ qh,
                                float* __restrict__ xsq,
                                float* __restrict__ tgtlogit, int B) {
  int b = blockIdx.x * blockDim.x + threadIdx.x;
  if (b >= B) return;
  const float* row = q + (size_t)b * 32;
  _Float16* out = qh + (size_t)b * 32;
  float xs = 0.f;
#pragma unroll
  for (int k = 0; k < 32; ++k) {
    float v = row[k];
    xs += v * v;
    out[k] = (_Float16)v;
  }
  xsq[b] = xs;

  int t = target[b];
  const float* crow = cand + (size_t)t * 32;
  float xy = 0.f, ys = 0.f;
#pragma unroll
  for (int k = 0; k < 32; ++k) {
    float cv = crow[k];
    xy += row[k] * cv;
    ys += cv * cv;
  }
  float A  = 1.f - 2.f * C_CURV * xy + C_CURV * ys;
  float Bc = 1.f - C_CURV * xs;
  float D  = 1.f - 2.f * C_CURV * xy + C_CURV2 * xs * ys;
  float num = A * A * xs - 2.f * A * Bc * xy + Bc * Bc * ys;
  float Dp = D + EPS_F;
  float dist = num * __builtin_amdgcn_rcpf(Dp * Dp);
  tgtlogit[b] = scaleP[0] * (marginP[0] - dist) + bias[t];
}

// ---------------------------------------------------------------------------
// Per-element logit from wmma xy and candidate metadata
// ---------------------------------------------------------------------------
__device__ __forceinline__ float murp_logit(float xy, float4 cm,
                                            float xqv, float cxqv,
                                            float Bcv, float Bc2v,
                                            float scale) {
  float u  = (2.f * C_CURV) * xy;          // 2c<q,y>
  float A  = cm.y - u;                     // 1 + c|y|^2 - 2c<q,y>
  float Q  = __builtin_fmaf(cxqv, cm.w, 1.f);
  float D  = Q - u;                        // 1 - 2c<q,y> + c^2|q|^2|y|^2
  float t1 = A * xqv;
  float t2 = Bcv * xy;
  float r  = Bc2v * cm.x;                  // Bc^2 |y|^2
  float h  = __builtin_fmaf(-(t2 + t2), A, r);
  float num = __builtin_fmaf(A, t1, h);
  float Dp = D + EPS_F;
  float dist = num * __builtin_amdgcn_rcpf(Dp * Dp);
  return __builtin_fmaf(-scale, dist, cm.z);   // scale*margin+bias - scale*dist
}

// ---------------------------------------------------------------------------
// Main: dual-WMMA tile GEMM + fused paired streaming logsumexp.
// grid = (B/16, NSLICES); block = 256 (8 waves). Each wave owns two 16x16
// candidate tiles per iteration; paired max update -> 3 exps per 2 logits.
// ---------------------------------------------------------------------------
__launch_bounds__(256)
__global__ void murp_main(const _Float16* __restrict__ qh,
                          const _Float16* __restrict__ candh,
                          const float* __restrict__ xsq,
                          const float4* __restrict__ cmeta,
                          const float* __restrict__ scaleP,
                          float* __restrict__ part_m,
                          float* __restrict__ part_s,
                          int B, int N) {
  __shared__ float lds_m[8][16];
  __shared__ float lds_s[8][16];

  const int tid   = threadIdx.x;
  const int wave  = tid >> 5;
  const int lane  = tid & 31;
  const int hi    = lane >> 4;     // half-wave selector
  const int lq    = lane & 15;     // row (A) / column (B,C,D) within tile
  const int rowBase = blockIdx.x * 16;
  const int slice   = blockIdx.y;

  const float scale = scaleP[0];

  // --- A fragment: 16 query rows x K=32, contiguous k = 16*hi + s (matches B) ---
  union { v16h v; v8h h[2]; } afrag;
  const _Float16* qrow = qh + (size_t)(rowBase + lq) * 32;
  afrag.h[0] = *(const v8h*)(qrow + 16 * hi);
  afrag.h[1] = *(const v8h*)(qrow + 16 * hi + 8);

  // Per-row-slot invariants
  float xq[8], cxq[8], Bc[8], Bc2[8];
#pragma unroll
  for (int v = 0; v < 8; ++v) {
    float x = xsq[rowBase + v + 8 * hi];
    xq[v]  = x;
    cxq[v] = C_CURV * x;
    Bc[v]  = 1.f - C_CURV * x;
    Bc2[v] = Bc[v] * Bc[v];
  }

  float m[8], s[8];
#pragma unroll
  for (int v = 0; v < 8; ++v) { m[v] = -3.0e38f; s[v] = 0.f; }

  const int nTiles = N >> 4;
  const int tps    = nTiles / NSLICES;
  const int tBeg   = slice * tps;
  const int tEnd   = tBeg + tps;
  const int tFull  = tBeg + (tps & ~15);   // pairs cover 16 tiles per sweep

  // --- Paired main loop: two tiles (n0, n0+16) per iteration ---
  for (int t = tBeg + 2 * wave; t < tFull; t += 16) {
    const int n0 = (t << 4) + lq;
    const int n1 = n0 + 16;
    v16h b0 = *(const v16h*)(candh + (size_t)n0 * 32 + 16 * hi);
    v16h b1 = *(const v16h*)(candh + (size_t)n1 * 32 + 16 * hi);
    float4 c0 = cmeta[n0];
    float4 c1 = cmeta[n1];

    v8f acc0 = {}, acc1 = {};
    acc0 = __builtin_amdgcn_wmma_f32_16x16x32_f16(
        false, afrag.v, false, b0, (short)0, acc0, false, false);
    acc1 = __builtin_amdgcn_wmma_f32_16x16x32_f16(
        false, afrag.v, false, b1, (short)0, acc1, false, false);

#pragma unroll
    for (int v = 0; v < 8; ++v) {
      float l0 = murp_logit(acc0[v], c0, xq[v], cxq[v], Bc[v], Bc2[v], scale);
      float l1 = murp_logit(acc1[v], c1, xq[v], cxq[v], Bc[v], Bc2[v], scale);
      float nm = fmaxf(m[v], fmaxf(l0, l1));
      s[v] = s[v] * __expf(m[v] - nm) + (__expf(l0 - nm) + __expf(l1 - nm));
      m[v] = nm;
    }
  }

  // --- Generic single-tile tail (tps not multiple of 16) ---
  for (int t = tFull + wave; t < tEnd; t += 8) {
    const int n = (t << 4) + lq;
    v16h b0 = *(const v16h*)(candh + (size_t)n * 32 + 16 * hi);
    float4 c0 = cmeta[n];
    v8f acc = {};
    acc = __builtin_amdgcn_wmma_f32_16x16x32_f16(
        false, afrag.v, false, b0, (short)0, acc, false, false);
#pragma unroll
    for (int v = 0; v < 8; ++v) {
      float l0 = murp_logit(acc[v], c0, xq[v], cxq[v], Bc[v], Bc2[v], scale);
      float nm = fmaxf(m[v], l0);
      s[v] = s[v] * __expf(m[v] - nm) + __expf(l0 - nm);
      m[v] = nm;
    }
  }

  // Reduce (m,s) across the 16 lanes holding the same row
#pragma unroll
  for (int off = 1; off < 16; off <<= 1) {
#pragma unroll
    for (int v = 0; v < 8; ++v) {
      float mo = __shfl_xor(m[v], off, 32);
      float so = __shfl_xor(s[v], off, 32);
      float nm = fmaxf(m[v], mo);
      s[v] = s[v] * __expf(m[v] - nm) + so * __expf(mo - nm);
      m[v] = nm;
    }
  }

  if (lq == 0) {
#pragma unroll
    for (int v = 0; v < 8; ++v) {
      lds_m[wave][v + 8 * hi] = m[v];
      lds_s[wave][v + 8 * hi] = s[v];
    }
  }
  __syncthreads();

  // Combine the 8 waves (fixed order -> deterministic), store slice partial
  if (tid < 16) {
    float mm = -3.0e38f, ss = 0.f;
#pragma unroll
    for (int w = 0; w < 8; ++w) {
      float mw = lds_m[w][tid], sw = lds_s[w][tid];
      float nm = fmaxf(mm, mw);
      ss = ss * __expf(mm - nm) + sw * __expf(mw - nm);
      mm = nm;
    }
    int row = rowBase + tid;
    part_m[(size_t)row * NSLICES + slice] = mm;
    part_s[(size_t)row * NSLICES + slice] = ss;
  }
}

// ---------------------------------------------------------------------------
// Per-row slice combine: lse(row) - tgt_logit(row)
// ---------------------------------------------------------------------------
__global__ void murp_combine(const float* __restrict__ part_m,
                             const float* __restrict__ part_s,
                             const float* __restrict__ tgtlogit,
                             float* __restrict__ rowres, int B) {
  int b = blockIdx.x * blockDim.x + threadIdx.x;
  if (b >= B) return;
  float mm = -3.0e38f, ss = 0.f;
#pragma unroll
  for (int sl = 0; sl < NSLICES; ++sl) {
    float mw = part_m[(size_t)b * NSLICES + sl];
    float sw = part_s[(size_t)b * NSLICES + sl];
    float nm = fmaxf(mm, mw);
    ss = ss * __expf(mm - nm) + sw * __expf(mw - nm);
    mm = nm;
  }
  rowres[b] = (mm + __logf(ss)) - tgtlogit[b];
}

// ---------------------------------------------------------------------------
// Deterministic final mean (single block, fixed-order tree reduce)
// ---------------------------------------------------------------------------
__global__ void murp_reduce(const float* __restrict__ rowres,
                            float* __restrict__ out, int B) {
  __shared__ float sm[256];
  float acc = 0.f;
  for (int i = threadIdx.x; i < B; i += 256) acc += rowres[i];
  sm[threadIdx.x] = acc;
  __syncthreads();
  for (int str = 128; str > 0; str >>= 1) {
    if ((int)threadIdx.x < str) sm[threadIdx.x] += sm[threadIdx.x + str];
    __syncthreads();
  }
  if (threadIdx.x == 0) out[0] = sm[0] / (float)B;
}

// ---------------------------------------------------------------------------
extern "C" void kernel_launch(void* const* d_in, const int* in_sizes, int n_in,
                              void* d_out, int out_size, void* d_ws, size_t ws_size,
                              hipStream_t stream) {
  const float* query   = (const float*)d_in[0];
  const float* cand    = (const float*)d_in[1];
  const int*   target  = (const int*)d_in[2];   // jnp.int64 demoted to int32 (x64 off)
  const float* bias    = (const float*)d_in[3];
  const float* scaleP  = (const float*)d_in[4];
  const float* marginP = (const float*)d_in[5];
  float* out = (float*)d_out;

  const int B = in_sizes[0] / 32;
  const int N = in_sizes[1] / 32;

  // Workspace carve-out (256B aligned slices)
  char* w = (char*)d_ws;
  auto alloc = [&](size_t bytes) -> char* {
    uintptr_t p = ((uintptr_t)w + 255) & ~(uintptr_t)255;
    char* r = (char*)p;
    w = r + bytes;
    return r;
  };
  _Float16* candh    = (_Float16*)alloc((size_t)N * 32 * sizeof(_Float16));
  _Float16* qh       = (_Float16*)alloc((size_t)B * 32 * sizeof(_Float16));
  float4*   cmeta    = (float4*)alloc((size_t)N * sizeof(float4));
  float*    xsq      = (float*)alloc((size_t)B * sizeof(float));
  float*    tgtlogit = (float*)alloc((size_t)B * sizeof(float));
  float*    part_m   = (float*)alloc((size_t)B * NSLICES * sizeof(float));
  float*    part_s   = (float*)alloc((size_t)B * NSLICES * sizeof(float));
  float*    rowres   = (float*)alloc((size_t)B * sizeof(float));

  murp_prep_cand<<<(N + 255) / 256, 256, 0, stream>>>(
      cand, bias, scaleP, marginP, candh, cmeta, N);
  murp_prep_query<<<(B + 255) / 256, 256, 0, stream>>>(
      query, cand, target, bias, scaleP, marginP, qh, xsq, tgtlogit, B);

  dim3 grid(B / 16, NSLICES);
  murp_main<<<grid, 256, 0, stream>>>(qh, candh, xsq, cmeta, scaleP,
                                      part_m, part_s, B, N);

  murp_combine<<<(B + 255) / 256, 256, 0, stream>>>(part_m, part_s, tgtlogit,
                                                    rowres, B);
  murp_reduce<<<1, 256, 0, stream>>>(rowres, out, B);
}
